// Transformer_47648367182578
// MI455X (gfx1250) — matrix operand order repaired and checked by reading, not compiled, not run on hardware
//
#include <hip/hip_runtime.h>

// ---------------------------------------------------------------------------
// HEPT-style transformer for MI455X (gfx1250, wave32).
// All GEMM-shaped math lowered to v_wmma_f32_16x16x32_f16 (f32 accumulate).
// LDS operand layouts chosen so every WMMA fragment load is ds_load_b128:
//   - A operands row-major  -> 2x b128 per lane
//   - B operands col-major  -> 2x b128 per lane (no u16 scatter, no packing)
// ---------------------------------------------------------------------------

typedef __attribute__((ext_vector_type(16))) _Float16 v16h;
typedef __attribute__((ext_vector_type(8)))  _Float16 v8h;
typedef __attribute__((ext_vector_type(8)))  float    v8f;

#define WMMA_F32_F16(a, b, c) \
  __builtin_amdgcn_wmma_f32_16x16x32_f16(false, (a), false, (b), (short)0, (c), false, false)

namespace hept {
constexpr int Bg   = 4;
constexpr int G    = 7472;
constexpr int N    = Bg * G;            // 29888
constexpr int BS   = 64;
constexpr int GP   = ((G + BS - 1) / BS) * BS;  // 7488
constexpr int NP   = Bg * GP;           // 29952
constexpr int NBLK = NP / BS;           // 468
constexpr int Ch   = 2;
constexpr int H    = 8;
constexpr int Dh   = 64;
constexpr int HD   = H * Dh;            // 512
constexpr int Lnum = 4;
constexpr int DO   = 32;
constexpr int ROWS = Ch * H;            // 16
constexpr int GCH  = (G + 255) / 256;   // 30 chunks per graph
constexpr int MB   = NP / 64;           // 468 M-blocks for all GEMMs (M == NP)
}

// ---------------------------------------------------------------------------
// WMMA fragment loaders (16x16x32 f16, wave32 lane layouts per CDNA5 ISA 7.12.2)
// ---------------------------------------------------------------------------
// A (16x32) from row-major f32 global, no bounds: 8x float2 (b64) loads.
__device__ __forceinline__ v16h ld_a32v(const float* __restrict__ A, int lda,
                                        int row0, int k0) {
  int lane = threadIdx.x & 31;
  int m = lane & 15, g = lane >> 4;
  const float* p = A + (row0 + m) * lda + k0 + g * 8;
  v16h r;
#pragma unroll
  for (int j = 0; j < 8; ++j) {
    int kk = ((j & 4) << 2) + (j & 3) * 2;   // {0|16} + 2*(j%4)
    float2 t = *(const float2*)(p + kk);
    r[2 * j] = (_Float16)t.x; r[2 * j + 1] = (_Float16)t.y;
  }
  return r;
}

// A (16x32) from row-major f16 LDS: two aligned v8h (ds_load_b128) loads.
__device__ __forceinline__ v16h ld_a16(const _Float16* A, int lda, int row0, int k0) {
  int lane = threadIdx.x & 31;
  int m = lane & 15, g = lane >> 4;
  const _Float16* p = A + (row0 + m) * lda + k0 + g * 8;
  v8h lo = *(const v8h*)p;
  v8h hi = *(const v8h*)(p + 16);
  return __builtin_shufflevector(lo, hi, 0, 1, 2, 3, 4, 5, 6, 7,
                                 8, 9, 10, 11, 12, 13, 14, 15);
}

// B (32x16) where storage is column-major: BT[n*ldk + k] == B[k][n].
// Each lane reads k = 8g..8g+7 and 16+8g..16+8g+7 (matches A) -> 2x ds_load_b128.
__device__ __forceinline__ v16h ld_bT(const _Float16* BT, int ldk, int k0, int col0) {
  int lane = threadIdx.x & 31;
  int n = lane & 15, g = lane >> 4;
  const _Float16* p = BT + (col0 + n) * ldk + k0 + g * 8;
  v8h lo = *(const v8h*)p;
  v8h hi = *(const v8h*)(p + 16);
  return __builtin_shufflevector(lo, hi, 0, 1, 2, 3, 4, 5, 6, 7,
                                 8, 9, 10, 11, 12, 13, 14, 15);
}

// ---------------------------------------------------------------------------
// Prep kernels
// ---------------------------------------------------------------------------
__global__ void zero16_kernel(int* p) {
  if (threadIdx.x < hept::ROWS) p[threadIdx.x] = 0;
}

// Pad fe_w1 (6x64) to 32x64 with zero rows so the GEMM needs no K bounds.
__global__ void padfe_kernel(const float* __restrict__ fe_w1, float* __restrict__ bpad) {
  int t = blockIdx.x * 256 + threadIdx.x;
  if (t >= 32 * 64) return;
  int k = t >> 6, n = t & 63;
  bpad[t] = (k < 6) ? fe_w1[k * 64 + n] : 0.f;
}

// Stable rank (matches jnp.argsort(argsort)) of eta/phi per graph, O(G^2) count.
__global__ __launch_bounds__(256) void grank_kernel(const float* __restrict__ coords,
                                                    int* __restrict__ rankE,
                                                    int* __restrict__ rankP) {
  using namespace hept;
  __shared__ float se[256], sp[256];
  int b = blockIdx.x / GCH, chunk = blockIdx.x % GCH;
  int li = chunk * 256 + threadIdx.x;
  bool act = li < G;
  int gi = b * G + li;
  float me = act ? coords[gi * 3 + 0] : 0.f;
  float mp = act ? coords[gi * 3 + 1] : 0.f;
  int ce = 0, cph = 0;
  for (int j0 = 0; j0 < G; j0 += 256) {
    int j = j0 + threadIdx.x;
    if (j < G) {
      se[threadIdx.x] = coords[(b * G + j) * 3 + 0];
      sp[threadIdx.x] = coords[(b * G + j) * 3 + 1];
    }
    __syncthreads();
    int lim = min(256, G - j0);
    if (act) {
      for (int t = 0; t < lim; ++t) {
        int j2 = j0 + t;
        float e = se[t], p = sp[t];
        ce  += (e < me) || (e == me && j2 < li);
        cph += (p < mp) || (p == mp && j2 < li);
      }
    }
    __syncthreads();
  }
  if (act) { rankE[gi] = ce; rankP[gi] = cph; }
}

// Region codes + per-row max (for the second _bit_shift).
__global__ void region1_kernel(const int* __restrict__ rankE, const int* __restrict__ rankP,
                               const int* __restrict__ regions, int* __restrict__ comb1,
                               int* __restrict__ rowmax) {
  using namespace hept;
  int idx = blockIdx.x * 256 + threadIdx.x;
  if (idx >= ROWS * N) return;
  int r = idx / N, i = idx % N;
  int c = r / H, h = r % H;
  int ec = regions[(c * 2 + 0) * H + h];
  int pc = regions[(c * 2 + 1) * H + h];
  float rsE = (float)G / (float)ec;
  float rsP = (float)G / (float)pc;
  int re = (int)ceilf((float)(rankE[i] + 1) / rsE);
  int rp = (int)ceilf((float)(rankP[i] + 1) / rsP);
  int nb = 32 - __clz(ec);                 // ceil(log2(max+1)); max(reg_eta) == ec
  int cb = (rp << nb) | re;
  comb1[idx] = cb;
  atomicMax(&rowmax[r], cb);
}

__global__ void region2_kernel(const int* __restrict__ comb1, const int* __restrict__ rowmax,
                               const int* __restrict__ batch, int* __restrict__ combU) {
  using namespace hept;
  int idx = blockIdx.x * 256 + threadIdx.x;
  if (idx >= ROWS * N) return;
  int r = idx / N, i = idx % N;
  int nb2 = 32 - __clz(rowmax[r]);
  combU[idx] = (batch[i] << nb2) | comb1[idx];
}

// Stable rank within graph of comb[0,0] (keys tie-broken by index).
__global__ __launch_bounds__(256) void padrank_kernel(const int* __restrict__ key,
                                                      int* __restrict__ rank0) {
  using namespace hept;
  __shared__ int sk[256];
  int b = blockIdx.x / GCH, chunk = blockIdx.x % GCH;
  int li = chunk * 256 + threadIdx.x;
  bool act = li < G;
  int gi = b * G + li;
  int mk = act ? key[gi] : 0;
  int cnt = 0;
  for (int j0 = 0; j0 < G; j0 += 256) {
    int j = j0 + threadIdx.x;
    if (j < G) sk[threadIdx.x] = key[b * G + j];
    __syncthreads();
    int lim = min(256, G - j0);
    if (act) {
      for (int t = 0; t < lim; ++t) {
        int j2 = j0 + t, k = sk[t];
        cnt += (k < mk) || (k == mk && j2 < li);
      }
    }
    __syncthreads();
  }
  if (act) rank0[gi] = cnt;
}

// pad_seq: main slots identity, fill slots = nodes ranked [G-64, G-48) per graph.
__global__ void padseq_kernel(const int* __restrict__ rank0, int* __restrict__ padseq) {
  using namespace hept;
  int i = blockIdx.x * 256 + threadIdx.x;
  if (i >= N) return;
  int b = i / G, gx = i % G;
  padseq[b * GP + gx] = i;
  int rk = rank0[i];
  int lo = G - BS;                 // 7408
  if (rk >= lo && rk < lo + (GP - G)) padseq[b * GP + G + (rk - lo)] = i;
}

// Gather padded x (zero-padded to 32 cols), coords, and the 16 key rows.
__global__ void gather_kernel(const float* __restrict__ x, const float* __restrict__ coords,
                              const int* __restrict__ combU, const int* __restrict__ padseq,
                              float* __restrict__ xp, float* __restrict__ cpad,
                              int* __restrict__ cs) {
  using namespace hept;
  int p = blockIdx.x * 256 + threadIdx.x;
  if (p >= NP) return;
  int s = padseq[p];
#pragma unroll
  for (int d = 0; d < 6; ++d) xp[p * 32 + d] = x[s * 6 + d];
#pragma unroll
  for (int d = 6; d < 32; ++d) xp[p * 32 + d] = 0.f;
#pragma unroll
  for (int d = 0; d < 3; ++d) cpad[p * 3 + d] = coords[s * 3 + d];
#pragma unroll
  for (int r = 0; r < ROWS; ++r) cs[r * NP + p] = combU[r * N + s];
}

// Stable rank of the 16 (hash,head) key rows over NP -> scatter into `order`.
__global__ __launch_bounds__(256) void orderrank_kernel(const int* __restrict__ cs,
                                                        int* __restrict__ order) {
  using namespace hept;
  __shared__ int sk[256];
  constexpr int CHUNKS = NP / 256;  // 117
  int r = blockIdx.x / CHUNKS, chunk = blockIdx.x % CHUNKS;
  int i = chunk * 256 + threadIdx.x;
  const int* row = cs + r * NP;
  int mk = row[i];
  int cnt = 0;
  for (int j0 = 0; j0 < NP; j0 += 256) {
    sk[threadIdx.x] = row[j0 + threadIdx.x];
    __syncthreads();
    for (int t = 0; t < 256; ++t) {
      int k = sk[t], j2 = j0 + t;
      cnt += (k < mk) || (k == mk && j2 < i);
    }
    __syncthreads();
  }
  order[r * NP + cnt] = i;
}

// w_head[l][h][c] = mean over (d, w) of Wr[l][(h*64+d)][c*8+w]
__global__ void whead_kernel(const float* __restrict__ wr, float* __restrict__ wh) {
  using namespace hept;
  int t = threadIdx.x;
  if (t >= Lnum * H * 2) return;
  int l = t / (H * 2);
  int h = (t >> 1) & 7;
  int cd = t & 1;
  const float* W = wr + l * HD * 16;
  float s = 0.f;
  for (int d = 0; d < 64; ++d)
    for (int w = 0; w < 8; ++w) s += W[(h * 64 + d) * 16 + cd * 8 + w];
  wh[t] = s * (1.0f / 512.0f);
}

// ---------------------------------------------------------------------------
// Generic WMMA GEMM: C = act(A@B + bias) + res.  M == NP always.
// Block = 4 waves = 64 rows x one 16-col tile; B panel staged in LDS
// COLUMN-MAJOR (Bst[n*K + k]) so every B fragment read is 2x ds_load_b128.
// Requires K % 32 == 0 (callers zero-pad), M % 64 == 0, N % 16 == 0.
// ---------------------------------------------------------------------------
template <int HAS_BIAS, int RELU, int HAS_RES>
__global__ __launch_bounds__(128) void gemm_kernel(
    const float* __restrict__ A, const float* __restrict__ Bm,
    const float* __restrict__ bias, const float* __restrict__ res,
    float* __restrict__ C, int K, int Nn,
    int lda, int ldb, int ldres, int ldc) {
  using namespace hept;
  __shared__ _Float16 Bst[16 * 512];     // col-major panel, up to K=512
  int wid = threadIdx.x >> 5;
  int lane = threadIdx.x & 31;
  int mb = blockIdx.x % MB;
  int ni = (blockIdx.x / MB) << 4;
  int mi = mb * 64 + wid * 16;

  // Fill Bst[n*K + k] = B[k][ni+n]: 16 coalesced b32 loads -> 2 b128 LDS stores.
  {
    int n = threadIdx.x & 15, seg = threadIdx.x >> 4;       // seg 0..7
    for (int k16 = seg; k16 < (K >> 4); k16 += 8) {
      const float* src = Bm + (k16 * 16) * ldb + ni + n;
      v8h p0, p1;
#pragma unroll
      for (int j = 0; j < 8; ++j) p0[j] = (_Float16)src[j * ldb];
#pragma unroll
      for (int j = 0; j < 8; ++j) p1[j] = (_Float16)src[(8 + j) * ldb];
      v8h* dst = (v8h*)(Bst + n * K + k16 * 16);
      dst[0] = p0; dst[1] = p1;
    }
  }
  __syncthreads();

  v8f acc = {};
  for (int k0 = 0; k0 < K; k0 += 32) {
    v16h a = ld_a32v(A, lda, mi, k0);
    v16h b = ld_bT(Bst, K, k0, 0);
    acc = WMMA_F32_F16(a, b, acc);
  }
  int n = lane & 15, g = lane >> 4;
  float bv = HAS_BIAS ? bias[ni + n] : 0.f;

  float rv[8];
  if (HAS_RES) {
#pragma unroll
    for (int r = 0; r < 8; ++r) rv[r] = res[(mi + r + 8 * g) * ldres + ni + n];
  }
#pragma unroll
  for (int r = 0; r < 8; ++r) {
    int row = mi + r + 8 * g;
    float v = acc[r] + bv;
    if (RELU) v = fmaxf(v, 0.f);
    if (HAS_RES) v += rv[r];
    C[row * ldc + ni + n] = v;
  }
}

// ---------------------------------------------------------------------------
// LayerNorm (+ optional tanh). One block per row, blockDim == row width.
// ---------------------------------------------------------------------------
__global__ void ln_kernel(const float* __restrict__ x, int ldx,
                          const float* __restrict__ gg, const float* __restrict__ bb,
                          float* __restrict__ y, int ldy, int act) {
  __shared__ float red[256];
  int row = blockIdx.x, t = threadIdx.x, d = blockDim.x;
  float v = x[row * ldx + t];
  red[t] = v; __syncthreads();
  for (int s = d >> 1; s > 0; s >>= 1) { if (t < s) red[t] += red[t + s]; __syncthreads(); }
  float mu = red[0] / (float)d;
  __syncthreads();
  float cv = v - mu;
  red[t] = cv * cv; __syncthreads();
  for (int s = d >> 1; s > 0; s >>= 1) { if (t < s) red[t] += red[t + s]; __syncthreads(); }
  float var = red[0] / (float)d;
  float o = cv * rsqrtf(var + 1e-5f) * gg[t] + bb[t];
  if (act) o = tanhf(o);
  y[row * ldy + t] = o;
}

// ---------------------------------------------------------------------------
// Fused block-local attention: one workgroup per (hash c, head h, 64-block).
// Qh/Kh row-major (A / transposed-B reads contiguous); Ws and V col-major
// (B reads contiguous; V's transposed store is a single packed b128/lane).
// ---------------------------------------------------------------------------
__global__ __launch_bounds__(256) void attn_kernel(
    const float* __restrict__ xn,                           // NP x 64 (LN'd)
    const float* __restrict__ Wq, const float* __restrict__ Wk,
    const float* __restrict__ Wv,                           // 64 x 512 (layer)
    const int* __restrict__ order,                          // 16 x NP
    const float* __restrict__ cp,                           // NP x 3
    const float* __restrict__ wh,                           // H x 2 (layer)
    _Float16* __restrict__ o0, _Float16* __restrict__ o1,   // NP x H x 64
    float* __restrict__ lse0, float* __restrict__ lse1)     // H x NP
{
  using namespace hept;
  __shared__ _Float16 Qh[64 * 96];   // row-major [token][feat]
  __shared__ _Float16 Kh[64 * 96];   // row-major [token][feat]
  __shared__ _Float16 Vt[64 * 64];   // col-major [dim][token]
  __shared__ _Float16 Xs[64 * 64];   // gathered features; reused as P (softmax)
  __shared__ float    Sf[64 * 64];   // scores f32; low half doubles as W tile
  __shared__ int      idxs[64];
  _Float16* Wst = reinterpret_cast<_Float16*>(Sf);  // col-major W tile; dead before Sf
  _Float16* Ps  = Xs;

  const int tid = threadIdx.x;
  const int lane = tid & 31, wid = tid >> 5;
  const int bid = blockIdx.x;
  const int c   = bid / (H * NBLK);
  const int h   = (bid / NBLK) % H;
  const int blk = bid % NBLK;

  if (tid < 64) idxs[tid] = order[(c * H + h) * NP + blk * 64 + tid];
  __syncthreads();

  // Gather 64 token rows (float4 loads, f16 LDS)
  for (int v = tid; v < 64 * 16; v += 256) {
    int r = v >> 4, c4 = (v & 15) << 2;
    float4 t = *(const float4*)(xn + idxs[r] * 64 + c4);
    _Float16* d = Xs + r * 64 + c4;
    d[0] = (_Float16)t.x; d[1] = (_Float16)t.y; d[2] = (_Float16)t.z; d[3] = (_Float16)t.w;
  }
  __syncthreads();

  // Q = Xs@Wq (scaled 1/8), K = Xs@Wk, V = Xs@Wv — 16 tiles over 8 waves each.
  for (int m = 0; m < 3; ++m) {
    const float* W = (m == 0) ? Wq : (m == 1) ? Wk : Wv;
    // Fill Wst[col*64 + k] = W[k][h*64+col]: strided b32 reads, 2 b128 stores.
    {
      int col = tid & 63, k16 = tid >> 6;                   // 256 threads exactly
      const float* src = W + (k16 * 16) * HD + h * 64 + col;
      v8h p0, p1;
#pragma unroll
      for (int j = 0; j < 8; ++j) p0[j] = (_Float16)src[j * HD];
#pragma unroll
      for (int j = 0; j < 8; ++j) p1[j] = (_Float16)src[(8 + j) * HD];
      v8h* dst = (v8h*)(Wst + col * 64 + k16 * 16);
      dst[0] = p0; dst[1] = p1;
    }
    __syncthreads();
#pragma unroll
    for (int tt = 0; tt < 2; ++tt) {
      int tile = wid * 2 + tt;
      int mi = (tile >> 2) * 16, ni = (tile & 3) * 16;
      v8f acc = {};
#pragma unroll
      for (int k0 = 0; k0 < 64; k0 += 32) {
        v16h a = ld_a16(Xs, 64, mi, k0);
        v16h b = ld_bT(Wst, 64, k0, ni);
        acc = WMMA_F32_F16(a, b, acc);
      }
      int n = lane & 15, g = lane >> 4;
      if (m == 2) {
        // V stored col-major: lane holds 8 consecutive rows of column ni+n.
        v8h pk;
#pragma unroll
        for (int r = 0; r < 8; ++r) pk[r] = (_Float16)acc[r];
        *(v8h*)(Vt + (ni + n) * 64 + mi + 8 * g) = pk;
      } else {
#pragma unroll
        for (int r = 0; r < 8; ++r) {
          int row = mi + r + 8 * g, col = ni + n;
          if (m == 0) Qh[row * 96 + col] = (_Float16)(acc[r] * 0.125f);  // D^-0.5
          else        Kh[row * 96 + col] = (_Float16)acc[r];
        }
      }
    }
    __syncthreads();
  }

  // Augmented dims: [sqrt2*a0, sqrt2*a1, -|a|^2, 1] (q) / [.., 1, -|a|^2] (k)
  if (tid < 64) {
    int tok = idxs[tid];
    float a0 = cp[tok * 3 + 0] * wh[h * 2 + 0];
    float a1 = cp[tok * 3 + 1] * wh[h * 2 + 1];
    float sq = a0 * a0 + a1 * a1;
    const float r2 = 1.41421356237f;
    Qh[tid * 96 + 64] = (_Float16)(r2 * a0); Qh[tid * 96 + 65] = (_Float16)(r2 * a1);
    Qh[tid * 96 + 66] = (_Float16)(-sq);     Qh[tid * 96 + 67] = (_Float16)1.0f;
    Kh[tid * 96 + 64] = (_Float16)(r2 * a0); Kh[tid * 96 + 65] = (_Float16)(r2 * a1);
    Kh[tid * 96 + 66] = (_Float16)1.0f;      Kh[tid * 96 + 67] = (_Float16)(-sq);
    for (int cc = 68; cc < 96; ++cc) {
      Qh[tid * 96 + cc] = (_Float16)0.f; Kh[tid * 96 + cc] = (_Float16)0.f;
    }
  }
  __syncthreads();

  // S = Qhat @ Khat^T  (K = 96, zero-padded); Kh row-major == col-major B.
#pragma unroll
  for (int tt = 0; tt < 2; ++tt) {
    int tile = wid * 2 + tt;
    int mi = (tile >> 2) * 16, ni = (tile & 3) * 16;
    v8f acc = {};
#pragma unroll
    for (int k0 = 0; k0 < 96; k0 += 32) {
      v16h a = ld_a16(Qh, 96, mi, k0);
      v16h b = ld_bT(Kh, 96, k0, ni);
      acc = WMMA_F32_F16(a, b, acc);
    }
    int n = lane & 15, g = lane >> 4;
#pragma unroll
    for (int r = 0; r < 8; ++r) Sf[(mi + r + 8 * g) * 64 + ni + n] = acc[r];
  }
  __syncthreads();

  // Row softmax + LSE
  if (tid < 64) {
    float mx = -3.0e38f;
    for (int j = 0; j < 64; ++j) mx = fmaxf(mx, Sf[tid * 64 + j]);
    float s = 0.f;
    for (int j = 0; j < 64; ++j) s += __expf(Sf[tid * 64 + j] - mx);
    float inv = 1.f / s;
    for (int j = 0; j < 64; ++j)
      Ps[tid * 64 + j] = (_Float16)(__expf(Sf[tid * 64 + j] - mx) * inv);
    float* lc = c ? lse1 : lse0;
    lc[h * NP + idxs[tid]] = mx + __logf(s);
  }
  __syncthreads();

  // O = P @ V, scatter to original padded positions; V read from col-major Vt.
  _Float16* oc = c ? o1 : o0;
#pragma unroll
  for (int tt = 0; tt < 2; ++tt) {
    int tile = wid * 2 + tt;
    int mi = (tile >> 2) * 16, ni = (tile & 3) * 16;
    v8f acc = {};
#pragma unroll
    for (int k0 = 0; k0 < 64; k0 += 32) {
      v16h a = ld_a16(Ps, 64, mi, k0);
      v16h b = ld_bT(Vt, 64, k0, ni);
      acc = WMMA_F32_F16(a, b, acc);
    }
    int n = lane & 15, g = lane >> 4;
#pragma unroll
    for (int r = 0; r < 8; ++r) {
      int row = mi + r + 8 * g;
      int pos = idxs[row];
      oc[(pos * H + h) * 64 + ni + n] = (_Float16)acc[r];
    }
  }
}

// Combine the two hash rounds with softmax over LSE.
__global__ void combine_kernel(const _Float16* __restrict__ o0, const _Float16* __restrict__ o1,
                               const float* __restrict__ lse0, const float* __restrict__ lse1,
                               float* __restrict__ ocomb) {
  using namespace hept;
  int t = blockIdx.x * 256 + threadIdx.x;
  if (t >= NP * HD) return;
  int tok = t >> 9, col = t & 511, h = col >> 6;
  float l0 = lse0[h * NP + tok], l1 = lse1[h * NP + tok];
  float m = fmaxf(l0, l1);
  float e0 = __expf(l0 - m), e1 = __expf(l1 - m);
  float inv = 1.f / (e0 + e1);
  ocomb[t] = ((float)o0[t] * e0 + (float)o1[t] * e1) * inv;
}

__global__ void unpad_kernel(const float* __restrict__ outp, float* __restrict__ out) {
  using namespace hept;
  int t = blockIdx.x * 256 + threadIdx.x;
  if (t >= N * DO) return;
  int node = t >> 5, d = t & 31;
  int b = node / G, gx = node % G;
  out[t] = outp[(b * GP + gx) * DO + d];
}

// ---------------------------------------------------------------------------
// Launcher
// ---------------------------------------------------------------------------
extern "C" void kernel_launch(void* const* d_in, const int* in_sizes, int n_in,
                              void* d_out, int out_size, void* d_ws, size_t ws_size,
                              hipStream_t stream) {
  using namespace hept;
  (void)in_sizes; (void)n_in; (void)out_size; (void)ws_size;

  const float* x      = (const float*)d_in[0];
  const float* coords = (const float*)d_in[1];
  const int*   batch  = (const int*)d_in[2];
  const int*   regions= (const int*)d_in[3];
  const float* fe_w1  = (const float*)d_in[4];
  const float* fe_b1  = (const float*)d_in[5];
  const float* fe_w2  = (const float*)d_in[6];
  const float* fe_b2  = (const float*)d_in[7];
  const float* ln1_g  = (const float*)d_in[8];
  const float* ln1_b  = (const float*)d_in[9];
  const float* wq     = (const float*)d_in[10];
  const float* wk     = (const float*)d_in[11];
  const float* wv     = (const float*)d_in[12];
  const float* wr     = (const float*)d_in[13];
  const float* wo     = (const float*)d_in[14];
  const float* bo     = (const float*)d_in[15];
  const float* ln2_g  = (const float*)d_in[16];
  const float* ln2_b  = (const float*)d_in[17];
  const float* ff_w1  = (const float*)d_in[18];
  const float* ff_b1  = (const float*)d_in[19];
  const float* ff_w2  = (const float*)d_in[20];
  const float* ff_b2  = (const float*)d_in[21];
  const float* Wproj  = (const float*)d_in[22];
  const float* m_w0   = (const float*)d_in[23];
  const float* m_b0   = (const float*)d_in[24];
  const float* m_wh   = (const float*)d_in[25];
  const float* m_bh   = (const float*)d_in[26];
  const float* m_wl   = (const float*)d_in[27];
  const float* m_bl   = (const float*)d_in[28];
  const float* m_lng  = (const float*)d_in[29];
  const float* m_lnb  = (const float*)d_in[30];
  float* out = (float*)d_out;

  // Bump allocator over d_ws (~195 MB high-water, deterministic layout).
  char* wp = (char*)d_ws;
  auto alloc = [&](size_t bytes) {
    char* p = wp;
    wp += (bytes + 255) & ~(size_t)255;
    return p;
  };
  int*      rankE  = (int*)alloc((size_t)N * 4);
  int*      rankP  = (int*)alloc((size_t)N * 4);
  int*      rank0  = (int*)alloc((size_t)N * 4);
  int*      rowmax = (int*)alloc((size_t)ROWS * 4);
  int*      comb1  = (int*)alloc((size_t)ROWS * N * 4);
  int*      combU  = (int*)alloc((size_t)ROWS * N * 4);
  int*      padseq = (int*)alloc((size_t)NP * 4);
  int*      cs     = (int*)alloc((size_t)ROWS * NP * 4);
  int*      order  = (int*)alloc((size_t)ROWS * NP * 4);
  float*    xp     = (float*)alloc((size_t)NP * 32 * 4);   // zero-padded to 32 cols
  float*    cpad   = (float*)alloc((size_t)NP * 3 * 4);
  float*    wh     = (float*)alloc((size_t)Lnum * H * 2 * 4);
  float*    fe1pad = (float*)alloc((size_t)32 * 64 * 4);   // fe_w1 padded to 32 rows
  float*    feats  = (float*)alloc((size_t)NP * Dh * (Lnum + 1) * 4);
  float*    xn     = (float*)alloc((size_t)NP * Dh * 4);
  float*    t1     = (float*)alloc((size_t)NP * Dh * 4);
  float*    lse0   = (float*)alloc((size_t)H * NP * 4);
  float*    lse1   = (float*)alloc((size_t)H * NP * 4);
  _Float16* o0     = (_Float16*)alloc((size_t)NP * HD * 2);
  _Float16* o1     = (_Float16*)alloc((size_t)NP * HD * 2);
  float*    ocomb  = (float*)alloc((size_t)NP * HD * 4);
  float*    ebuf   = (float*)alloc((size_t)NP * DO * 4);
  float*    m1 = (float*)o0;   // NP*256 f32 fits exactly in o0's bytes (free post-attn)
  float*    m2 = (float*)o1;
  float*    outp = ocomb;      // NP*32 f32 (free post-attn)

  // ---- prepare ----
  zero16_kernel<<<1, 32, 0, stream>>>(rowmax);
  padfe_kernel<<<dim3(8), dim3(256), 0, stream>>>(fe_w1, fe1pad);
  grank_kernel<<<dim3(Bg * GCH), dim3(256), 0, stream>>>(coords, rankE, rankP);
  region1_kernel<<<dim3((ROWS * N + 255) / 256), dim3(256), 0, stream>>>(rankE, rankP, regions,
                                                                         comb1, rowmax);
  region2_kernel<<<dim3((ROWS * N + 255) / 256), dim3(256), 0, stream>>>(comb1, rowmax, batch,
                                                                         combU);
  padrank_kernel<<<dim3(Bg * GCH), dim3(256), 0, stream>>>(combU, rank0);
  padseq_kernel<<<dim3((N + 255) / 256), dim3(256), 0, stream>>>(rank0, padseq);
  gather_kernel<<<dim3(NP / 256), dim3(256), 0, stream>>>(x, coords, combU, padseq, xp, cpad, cs);
  orderrank_kernel<<<dim3(ROWS * (NP / 256)), dim3(256), 0, stream>>>(cs, order);
  whead_kernel<<<1, 64, 0, stream>>>(wr, wh);

  // GEMM dispatch: (bias, relu, res) combinations are compile-time.
  auto gemm_br  = [&](const float* A, int lda, const float* Bm, int ldb, const float* bias,
                      float* Cm, int ldc, int K, int Nn) {   // bias + relu
    gemm_kernel<1, 1, 0><<<dim3(MB * (Nn / 16)), dim3(128), 0, stream>>>(
        A, Bm, bias, nullptr, Cm, K, Nn, lda, ldb, 0, ldc);
  };
  auto gemm_b   = [&](const float* A, int lda, const float* Bm, int ldb, const float* bias,
                      float* Cm, int ldc, int K, int Nn) {   // bias only
    gemm_kernel<1, 0, 0><<<dim3(MB * (Nn / 16)), dim3(128), 0, stream>>>(
        A, Bm, bias, nullptr, Cm, K, Nn, lda, ldb, 0, ldc);
  };
  auto gemm_bres= [&](const float* A, int lda, const float* Bm, int ldb, const float* bias,
                      const float* res, int ldres, float* Cm, int ldc, int K, int Nn) {
    gemm_kernel<1, 0, 1><<<dim3(MB * (Nn / 16)), dim3(128), 0, stream>>>(
        A, Bm, bias, res, Cm, K, Nn, lda, ldb, ldres, ldc);
  };
  auto gemm_nn  = [&](const float* A, int lda, const float* Bm, int ldb,
                      float* Cm, int ldc, int K, int Nn) {   // plain
    gemm_kernel<0, 0, 0><<<dim3(MB * (Nn / 16)), dim3(128), 0, stream>>>(
        A, Bm, nullptr, nullptr, Cm, K, Nn, lda, ldb, 0, ldc);
  };

  // ---- feature embedding -> feats[:, 0:64] ----
  gemm_br(xp, 32, fe1pad, 64, fe_b1, t1, 64, 32, 64);
  gemm_b (t1, 64, fe_w2, 64, fe_b2, feats, 320, 64, 64);

  // ---- 4 attention layers ----
  for (int l = 0; l < Lnum; ++l) {
    const float* xcur = feats + l * 64;
    float* xnext = feats + (l + 1) * 64;
    ln_kernel<<<dim3(NP), dim3(64), 0, stream>>>(xcur, 320, ln1_g + l * 64, ln1_b + l * 64,
                                                 xn, 64, 0);
    attn_kernel<<<dim3(Ch * H * NBLK), dim3(256), 0, stream>>>(
        xn, wq + l * Dh * HD, wk + l * Dh * HD, wv + l * Dh * HD,
        order, cpad, wh + l * H * 2, o0, o1, lse0, lse1);
    combine_kernel<<<dim3(NP * HD / 256), dim3(256), 0, stream>>>(o0, o1, lse0, lse1, ocomb);
    gemm_bres(ocomb, 512, wo + l * HD * Dh, 64, bo + l * 64, xcur, 320, xnext, 320, 512, 64);
    ln_kernel<<<dim3(NP), dim3(64), 0, stream>>>(xnext, 320, ln2_g + l * 64, ln2_b + l * 64,
                                                 xn, 64, 0);
    gemm_br(xn, 64, ff_w1 + l * Dh * Dh, 64, ff_b1 + l * 64, t1, 64, 64, 64);
    gemm_bres(t1, 64, ff_w2 + l * Dh * Dh, 64, ff_b2 + l * 64, xnext, 320, xnext, 320, 64, 64);
  }

  // ---- projection + MLP head ----
  gemm_nn(feats, 320, Wproj, 32, ebuf, 32, 320, 32);
  gemm_b (ebuf, 32, m_w0, 256, m_b0, m1, 256, 32, 256);
  ln_kernel<<<dim3(NP), dim3(256), 0, stream>>>(m1, 256, m_lng, m_lnb, m2, 256, 1);
  for (int i = 0; i < 3; ++i) {
    gemm_b(m2, 256, m_wh + i * 256 * 256, 256, m_bh + i * 256, m1, 256, 256, 256);
    ln_kernel<<<dim3(NP), dim3(256), 0, stream>>>(m1, 256, m_lng + (i + 1) * 256,
                                                  m_lnb + (i + 1) * 256, m2, 256, 1);
  }
  gemm_bres(m2, 256, m_wl, 32, m_bl, ebuf, 32, outp, 32, 256, 32);
  unpad_kernel<<<dim3(N * DO / 256), dim3(256), 0, stream>>>(outp, out);
}
